// SPUPMDNet_5866925326381
// MI455X (gfx1250) — matrix operand order repaired and checked by reading, not compiled, not run on hardware
//
#include <hip/hip_runtime.h>
#include <math.h>

// ---------------------------------------------------------------------------
// SPU-PMD-style refinement net on MI455X (gfx1250, wave32).
// 1x1 convs -> v_wmma_f32_16x16x32_f16, 64x64 tile, 2 WMMA/wave/K-step,
// double-buffered LDS pipeline, b128 global loads, b64 LDS stores.
// Glue kernels fully float4-vectorized (net is HBM-bandwidth bound).
// ---------------------------------------------------------------------------

#define B_   4
#define N_   4096
#define KNN_ 16
#define NB_  (B_ * N_)

typedef _Float16 v16h __attribute__((ext_vector_type(16)));
typedef _Float16 v8h  __attribute__((ext_vector_type(8)));
typedef _Float16 v4h  __attribute__((ext_vector_type(4)));
typedef float    v8f  __attribute__((ext_vector_type(8)));
typedef float    v4f  __attribute__((ext_vector_type(4)));
typedef int      v4i  __attribute__((ext_vector_type(4)));

// ========================= WMMA 1x1-conv GEMM ==============================
// y[b, o, m] = act( sum_c w[o,c] * x[b,c,m] + bias[o] )
// Block: 256 thr = 8 waves (2x4); tile 64(Cout) x 64(M); wave: two 16x16 WMMA.
template<bool KPAD>
__global__ void __launch_bounds__(256) conv1x1_wmma(
    const float* __restrict__ x, const float* __restrict__ w,
    const float* __restrict__ bias, float* __restrict__ y,
    int Cin, int Cout, int M, int act)
{
    __shared__ __align__(16) _Float16 As[2][64 * 40];   // [m][k], pitch 40
    __shared__ __align__(16) _Float16 Bs[2][32 * 72];   // [k][n], pitch 72

    const int t  = threadIdx.x;
    const int m0 = blockIdx.x * 64;            // spatial base (M % 64 == 0)
    const int c0 = blockIdx.y * 64;            // cout base
    const float* xb = x + (size_t)blockIdx.z * (size_t)Cin  * (size_t)M;
    float*       yb = y + (size_t)blockIdx.z * (size_t)Cout * (size_t)M;

    const int wid = t >> 5, lane = t & 31;
    const int wm  = wid & 1, wn = wid >> 1;    // 2(m) x 4(n) wave grid
    const int hh  = lane >> 4, lr = lane & 15;

    v4f aR[2], bR[2];                          // staged registers (pipeline)

    auto gload = [&](int k0) {
        #pragma unroll
        for (int i = 0; i < 2; ++i) {
            int id = t + 256 * i;
            int mm = id >> 3, kb = (id & 7) * 4;
            int row = c0 + mm;
            int rc  = row < Cout ? row : Cout - 1;
            if (KPAD) {
                #pragma unroll
                for (int j = 0; j < 4; ++j) {
                    int kk = k0 + kb + j;
                    int kc = kk < Cin ? kk : 0;
                    float v = w[(size_t)rc * Cin + kc];
                    aR[i][j] = (row < Cout && kk < Cin) ? v : 0.f;
                }
            } else {
                v4f v = *(const v4f*)(w + (size_t)rc * Cin + (k0 + kb));
                aR[i] = (row < Cout) ? v : (v4f){0.f, 0.f, 0.f, 0.f};
            }
        }
        #pragma unroll
        for (int i = 0; i < 2; ++i) {
            int id = t + 256 * i;
            int kk = id >> 4, nb = (id & 15) * 4;
            int kr = k0 + kk;
            if (KPAD) {
                int kc = kr < Cin ? kr : 0;
                #pragma unroll
                for (int j = 0; j < 4; ++j) {
                    float v = xb[(size_t)kc * M + (m0 + nb + j)];
                    bR[i][j] = (kr < Cin) ? v : 0.f;
                }
            } else {
                bR[i] = *(const v4f*)(xb + (size_t)kr * M + (m0 + nb));
            }
        }
    };
    auto lstore = [&](int buf) {
        #pragma unroll
        for (int i = 0; i < 2; ++i) {
            int id = t + 256 * i;
            int mm = id >> 3, kb = (id & 7) * 4;
            *(v4h*)&As[buf][mm * 40 + kb] = __builtin_convertvector(aR[i], v4h);
        }
        #pragma unroll
        for (int i = 0; i < 2; ++i) {
            int id = t + 256 * i;
            int kk = id >> 4, nb = (id & 15) * 4;
            *(v4h*)&Bs[buf][kk * 72 + nb] = __builtin_convertvector(bR[i], v4h);
        }
    };

    const int nk = (Cin + 31) >> 5;
    gload(0);
    lstore(0);

    v8f acc0 = {0.f,0.f,0.f,0.f,0.f,0.f,0.f,0.f};
    v8f acc1 = {0.f,0.f,0.f,0.f,0.f,0.f,0.f,0.f};

    for (int s = 0; s < nk; ++s) {
        __syncthreads();                        // buf (s&1) ready for all waves
        if (s + 1 < nk) {
            gload((s + 1) * 32);                // issue global loads, no wait yet
            if (s + 2 < nk)                     // L2 prefetch two slabs ahead
                __builtin_prefetch(xb + (size_t)(s + 2) * 32 * M + m0 + lr, 0, 1);
        }
        const int buf = s & 1;

        // ---- B fragment: n = wn*16+lr ; k = 16*(e>>3)+(e&7)+8*hh (gather, row-major LDS)
        v16h bf;
        const int bcol = wn * 16 + lr;
        #pragma unroll
        for (int e = 0; e < 16; ++e)
            bf[e] = Bs[buf][(16 * (e >> 3) + (e & 7) + 8 * hh) * 72 + bcol];

        // ---- A fragments: m = j*32 + wm*16+lr ; k = 16*(e>>3)+(e&7)+8*hh
        {
            const int ar0 = (wm * 16 + lr) * 40;
            v8h alo = *(const v8h*)&As[buf][ar0 + 8 * hh];
            v8h ahi = *(const v8h*)&As[buf][ar0 + 16 + 8 * hh];
            v16h a = __builtin_shufflevector(alo, ahi,
                     0,1,2,3,4,5,6,7,8,9,10,11,12,13,14,15);
            acc0 = __builtin_amdgcn_wmma_f32_16x16x32_f16(
                       false, a, false, bf, (short)0, acc0, false, false);
        }
        {
            const int ar1 = (32 + wm * 16 + lr) * 40;
            v8h alo = *(const v8h*)&As[buf][ar1 + 8 * hh];
            v8h ahi = *(const v8h*)&As[buf][ar1 + 16 + 8 * hh];
            v16h a = __builtin_shufflevector(alo, ahi,
                     0,1,2,3,4,5,6,7,8,9,10,11,12,13,14,15);
            acc1 = __builtin_amdgcn_wmma_f32_16x16x32_f16(
                       false, a, false, bf, (short)0, acc1, false, false);
        }

        if (s + 1 < nk) lstore((s + 1) & 1);    // cvt waits on loads AFTER wmma
    }

    // ---- epilogue: D vgpr r -> m = r + 8*hh, n = lr ----
    const int nc = m0 + wn * 16 + lr;
    #pragma unroll
    for (int r = 0; r < 8; ++r) {
        int mr = c0 + wm * 16 + r + 8 * hh;
        if (mr < Cout) {
            float v = acc0[r] + bias[mr];
            if (act == 1) v = v > 0.f ? v : 0.f;
            yb[(size_t)mr * M + nc] = v;
        }
        int mr2 = mr + 32;
        if (mr2 < Cout) {
            float v = acc1[r] + bias[mr2];
            if (act == 1) v = v > 0.f ? v : 0.f;
            yb[(size_t)mr2 * M + nc] = v;
        }
    }
}

// ========================= KNN-16 (B,3,N) ==================================
__global__ void __launch_bounds__(256) knn16(const float* __restrict__ pts,
                                             int* __restrict__ idx)
{
    __shared__ float tx[256], ty[256], tz[256];
    const int b = blockIdx.y;
    const int n = blockIdx.x * 256 + threadIdx.x;
    const float* pb = pts + (size_t)b * 3 * N_;
    const float qx = pb[n], qy = pb[N_ + n], qz = pb[2 * N_ + n];

    float dist[KNN_]; int ind[KNN_];
    #pragma unroll
    for (int i = 0; i < KNN_; ++i) { dist[i] = 3.4e38f; ind[i] = 0; }

    for (int t = 0; t < N_; t += 256) {
        __syncthreads();
        tx[threadIdx.x] = pb[t + threadIdx.x];
        ty[threadIdx.x] = pb[N_ + t + threadIdx.x];
        tz[threadIdx.x] = pb[2 * N_ + t + threadIdx.x];
        __syncthreads();
        for (int j = 0; j < 256; ++j) {
            float dx = qx - tx[j], dy = qy - ty[j], dz = qz - tz[j];
            float d = dx * dx + dy * dy + dz * dz;
            if (d < dist[KNN_ - 1]) {            // fully-unrolled reg sort
                dist[KNN_ - 1] = d; ind[KNN_ - 1] = t + j;
                #pragma unroll
                for (int q = KNN_ - 1; q > 0; --q)
                    if (dist[q] < dist[q - 1]) {
                        float td = dist[q]; dist[q] = dist[q-1]; dist[q-1] = td;
                        int   ti = ind[q];  ind[q]  = ind[q-1];  ind[q-1]  = ti;
                    }
            }
        }
    }
    int* ob = idx + ((size_t)b * N_ + n) * KNN_;
    #pragma unroll
    for (int i = 0; i < KNN_; ++i) ob[i] = ind[i];
}

// ==================== BatchNorm (training batch stats) =====================
// Msp = 1<<mshift (4096 or 65536); C power of two. v4f loads, shift/mask math.
__global__ void __launch_bounds__(256) bn_stats(const float* __restrict__ x,
    float* __restrict__ mean, float* __restrict__ rstd, int C, int mshift, float eps)
{
    __shared__ float ss[256], sq[256];
    const int c = blockIdx.x;
    const long Msp = 1L << mshift;
    const long per = (long)B_ << mshift;
    float s = 0.f, s2 = 0.f;
    for (long i = (long)threadIdx.x * 4; i < per; i += 256 * 4) {
        long b = i >> mshift, m = i & (Msp - 1);
        v4f v = *(const v4f*)&x[(((size_t)b * C + c) << mshift) + m];
        s  += v[0] + v[1] + v[2] + v[3];
        s2 += v[0]*v[0] + v[1]*v[1] + v[2]*v[2] + v[3]*v[3];
    }
    ss[threadIdx.x] = s; sq[threadIdx.x] = s2; __syncthreads();
    for (int o = 128; o > 0; o >>= 1) {
        if (threadIdx.x < o) {
            ss[threadIdx.x] += ss[threadIdx.x + o];
            sq[threadIdx.x] += sq[threadIdx.x + o];
        }
        __syncthreads();
    }
    if (threadIdx.x == 0) {
        float m = ss[0] / (float)per;
        float v = sq[0] / (float)per - m * m;
        if (v < 0.f) v = 0.f;
        mean[c] = m; rstd[c] = rsqrtf(v + eps);
    }
}

__global__ void bn_act4(float* __restrict__ y, const float* __restrict__ x,
    const float* __restrict__ mean, const float* __restrict__ rstd,
    const float* __restrict__ g, const float* __restrict__ bb,
    int Cmask, int mshift, long total4, int act)
{
    long i = (long)blockIdx.x * 256 + threadIdx.x;
    if (i >= total4) return;
    long i4 = i * 4;
    int c = (int)((i4 >> mshift) & Cmask);
    float sc = g[c] * rstd[c], mu = mean[c], bc = bb[c];
    v4f v = *(const v4f*)&x[i4];
    #pragma unroll
    for (int j = 0; j < 4; ++j) {
        float u = sc * (v[j] - mu) + bc;
        if (act == 1) u = u > 0.f ? u : 0.f;
        else if (act == 2) u = 1.f / (1.f + __expf(-u));
        v[j] = u;
    }
    *(v4f*)&y[i4] = v;
}

// ================== elementwise glue (float4 vectorized) ===================
__global__ void ew_add4(float* o, const float* a, const float* b, long n4) {
    long i = (long)blockIdx.x * 256 + threadIdx.x;
    if (i < n4) *(v4f*)&o[i*4] = *(const v4f*)&a[i*4] + *(const v4f*)&b[i*4];
}
__global__ void ew_mul4(float* o, const float* a, const float* b, long n4) {
    long i = (long)blockIdx.x * 256 + threadIdx.x;
    if (i < n4) *(v4f*)&o[i*4] = *(const v4f*)&a[i*4] * *(const v4f*)&b[i*4];
}
__global__ void gru_mix4(float* feat, const float* z, const float* hh, long n4) {
    long i = (long)blockIdx.x * 256 + threadIdx.x;
    if (i >= n4) return;
    v4f zz = *(const v4f*)&z[i*4];
    v4f cu = *(const v4f*)&feat[i*4];
    v4f hv = *(const v4f*)&hh[i*4];
    v4f one = {1.f, 1.f, 1.f, 1.f};
    *(v4f*)&feat[i*4] = (one - zz) * cu + zz * hv;
}

// out[b,c,n,k0..k0+3] = q[b,c,n] - src[b,c,idx[b,n,k]]
__global__ void sub_group4(float* __restrict__ out, const float* __restrict__ q,
                           const float* __restrict__ src, const int* __restrict__ idx,
                           int C, long total4)
{
    long i = (long)blockIdx.x * 256 + threadIdx.x;
    if (i >= total4) return;
    long i4 = i * 4;
    int k = (int)(i4 & 15);
    int n = (int)((i4 >> 4) & (N_ - 1));
    int c = (int)((i4 >> 16) % C);
    int b = (int)( i4 / (16L * N_ * C));
    v4i jj = *(const v4i*)&idx[((size_t)b * N_ + n) * 16 + k];
    size_t base = ((size_t)b * C + c) * N_;
    float qv = q[base + n];
    v4f o;
    #pragma unroll
    for (int j = 0; j < 4; ++j) o[j] = qv - src[base + jj[j]];
    *(v4f*)&out[i4] = o;
}

// out[b,c,n,k0..k0+3] = a[b,c,n] + pe[b,c,n,k]
__global__ void badd_group4(float* __restrict__ out, const float* __restrict__ a,
                            const float* __restrict__ pe, int C, long total4)
{
    long i = (long)blockIdx.x * 256 + threadIdx.x;
    if (i >= total4) return;
    long i4 = i * 4;
    int n = (int)((i4 >> 4) & (N_ - 1));
    int c = (int)((i4 >> 16) % C);
    int b = (int)( i4 / (16L * N_ * C));
    float av = a[((size_t)b * C + c) * N_ + n];
    v4f p = *(const v4f*)&pe[i4];
    v4f o = {av + p[0], av + p[1], av + p[2], av + p[3]};
    *(v4f*)&out[i4] = o;
}

__global__ void softmax16(float* __restrict__ x, long rows) {
    long r = (long)blockIdx.x * 256 + threadIdx.x;
    if (r >= rows) return;
    v4f* p = (v4f*)(x + r * 16);
    v4f v0 = p[0], v1 = p[1], v2 = p[2], v3 = p[3];
    float m = v0[0];
    #pragma unroll
    for (int j = 0; j < 4; ++j) {
        m = v0[j] > m ? v0[j] : m; m = v1[j] > m ? v1[j] : m;
        m = v2[j] > m ? v2[j] : m; m = v3[j] > m ? v3[j] : m;
    }
    float s = 0.f;
    #pragma unroll
    for (int j = 0; j < 4; ++j) {
        v0[j] = __expf(v0[j] - m); s += v0[j];
        v1[j] = __expf(v1[j] - m); s += v1[j];
        v2[j] = __expf(v2[j] - m); s += v2[j];
        v3[j] = __expf(v3[j] - m); s += v3[j];
    }
    float inv = 1.f / s;
    v4f vi = {inv, inv, inv, inv};
    p[0] = v0 * vi; p[1] = v1 * vi; p[2] = v2 * vi; p[3] = v3 * vi;
}

__global__ void attn_agg(float* __restrict__ agg, const float* __restrict__ attn,
                         const float* __restrict__ val, long rows) {
    long r = (long)blockIdx.x * 256 + threadIdx.x;
    if (r >= rows) return;
    const v4f* pa = (const v4f*)(attn + r * 16);
    const v4f* pv = (const v4f*)(val + r * 16);
    float s = 0.f;
    #pragma unroll
    for (int j = 0; j < 4; ++j) {
        v4f a = pa[j], v = pv[j];
        s += a[0]*v[0] + a[1]*v[1] + a[2]*v[2] + a[3]*v[3];
    }
    agg[r] = s;
}

__global__ void __launch_bounds__(256) maxpool_n(const float* __restrict__ x,
                                                 float* __restrict__ out, int C) {
    __shared__ float red[256];
    int c = blockIdx.x, b = blockIdx.y;
    const float* p = x + ((size_t)b * C + c) * N_;
    float m = -3.4e38f;
    for (int n = threadIdx.x * 4; n < N_; n += 256 * 4) {
        v4f v = *(const v4f*)&p[n];
        #pragma unroll
        for (int j = 0; j < 4; ++j) m = v[j] > m ? v[j] : m;
    }
    red[threadIdx.x] = m; __syncthreads();
    for (int o = 128; o > 0; o >>= 1) {
        if (threadIdx.x < o && red[threadIdx.x + o] > red[threadIdx.x])
            red[threadIdx.x] = red[threadIdx.x + o];
        __syncthreads();
    }
    if (threadIdx.x == 0) out[b * C + c] = red[0];
}

// out (B,256,N) = [a(B,128,N) | b];  bcast: b is (B,128) broadcast over N
__global__ void concat256_4(float* __restrict__ o, const float* __restrict__ a,
                            const float* __restrict__ b, int bcast, long total4)
{
    long i = (long)blockIdx.x * 256 + threadIdx.x;
    if (i >= total4) return;
    long i4 = i * 4;
    int n  = (int)(i4 & (N_ - 1));
    int c  = (int)((i4 >> 12) & 255);
    int bi = (int)( i4 >> 20);               // / (256*4096)
    if (c < 128) {
        *(v4f*)&o[i4] = *(const v4f*)&a[(((size_t)bi * 128 + c) << 12) + n];
    } else if (bcast) {
        float v = b[(size_t)bi * 128 + (c - 128)];
        v4f s = {v, v, v, v};
        *(v4f*)&o[i4] = s;
    } else {
        *(v4f*)&o[i4] = *(const v4f*)&b[(((size_t)bi * 128 + (c - 128)) << 12) + n];
    }
}

// p_out = p_in + rate * 0.5*(sign(gate)+1) * tanh(reg)
__global__ void gcr_apply(float* __restrict__ pout, const float* __restrict__ pin,
                          const float* __restrict__ gate, const float* __restrict__ reg,
                          float rate)
{
    int i = blockIdx.x * 256 + threadIdx.x;
    if (i >= NB_) return;
    int b = i / N_, n = i & (N_ - 1);
    float gv = gate[i];
    float g = gv > 0.f ? 1.f : (gv < 0.f ? 0.f : 0.5f);
    #pragma unroll
    for (int d = 0; d < 3; ++d) {
        size_t o = ((size_t)b * 3 + d) * N_ + n;
        pout[o] = pin[o] + rate * g * tanhf(reg[o]);
    }
}

// ============================ host orchestration ===========================
static inline int cdiv(long a, long b) { return (int)((a + b - 1) / b); }

static void conv(const float* x, const float* w, const float* b, float* y,
                 int Cin, int Cout, long M, int act, hipStream_t s) {
    dim3 g(cdiv(M, 64), cdiv(Cout, 64), B_);
    if ((Cin & 31) == 0)
        conv1x1_wmma<false><<<g, 256, 0, s>>>(x, w, b, y, Cin, Cout, (int)M, act);
    else
        conv1x1_wmma<true><<<g, 256, 0, s>>>(x, w, b, y, Cin, Cout, (int)M, act);
}
static void bn(float* x, int C, int mshift, const float* g, const float* bb,
               float* bnm, float* bnr, int act, hipStream_t s) {
    bn_stats<<<C, 256, 0, s>>>(x, bnm, bnr, C, mshift, 1e-5f);
    long total4 = ((long)B_ * C << mshift) >> 2;
    bn_act4<<<cdiv(total4, 256), 256, 0, s>>>(x, x, bnm, bnr, g, bb,
                                              C - 1, mshift, total4, act);
}

struct WS {
    float *t64, *f1, *gmax, *f2, *t256, *f3, *x64, *keyf, *qf, *vf;
    float *posrel, *peraw, *pe, *qk, *a1, *agg, *g64, *g16, *g1, *r3;
    float *zb, *rb, *hhb, *bnm, *bnr;
    int* idx;
};

// Param leaf order inside one stage (jax pytree: dict keys sorted, lists kept,
// {'b','w'}->b,w ; {'b','g'}->b,g):
//  0..5  gcr_g[0..2]{b,w} ; 6..9 gcr_reg[0..1]{b,w}
// 10..13 mlp1[0..1]{b,w}  ; 14..17 mlp2[0..1]{b,w}
// [rfa: 18..29 h{bn{b,g},conv{b,w}}, r{...}, z{...}]
// tr (sorted): attn1, attn2, attn_bn, end, key, pos1, pos2, pos_bn, query,
//              start, value -> {b,w} / {b,g} pairs.
static void run_stage(const float* pts, float* pts_out, float* feat,
                      const float* pre_h, const float* const* P, int with_rfa,
                      const WS& w, hipStream_t s)
{
    const int TR = with_rfa ? 30 : 18;
    const long NK = (long)N_ * 16;          // per-batch spatial for 4D convs
    const long E64 = 64L * B_ * NK;         // elems of (B,64,N,16)

    // ---- extractor: mlp1, maxpool-concat, mlp2 ----
    conv(pts,    P[11], P[10], w.t64, 3,   64,  N_, 1, s);
    conv(w.t64,  P[13], P[12], w.f1,  64,  128, N_, 0, s);
    maxpool_n<<<dim3(128, B_), 256, 0, s>>>(w.f1, w.gmax, 128);
    concat256_4<<<cdiv(64L * NB_, 256), 256, 0, s>>>(w.f2, w.f1, w.gmax, 1, 64L * NB_);
    conv(w.f2,   P[15], P[14], w.t256, 256, 256, N_, 1, s);
    conv(w.t256, P[17], P[16], w.f3,   256, 128, N_, 0, s);

    // ---- transformer ----
    conv(w.f3, P[TR+19], P[TR+18], w.x64, 128, 64, N_, 0, s);      // start
    knn16<<<dim3(N_ / 256, B_), 256, 0, s>>>(pts, w.idx);
    conv(w.x64, P[TR+9],  P[TR+8],  w.keyf, 64, 64, N_, 0, s);     // key
    conv(w.x64, P[TR+17], P[TR+16], w.qf,   64, 64, N_, 0, s);     // query
    conv(w.x64, P[TR+21], P[TR+20], w.vf,   64, 64, N_, 0, s);     // value
    sub_group4<<<cdiv(E64/4, 256), 256, 0, s>>>(w.qk, w.qf, w.keyf, w.idx, 64, E64/4);
    sub_group4<<<cdiv(3L*B_*NK/4, 256), 256, 0, s>>>(w.posrel, pts, pts, w.idx, 3, 3L*B_*NK/4);
    conv(w.posrel, P[TR+11], P[TR+10], w.peraw, 3, 64, NK, 0, s);  // pos1
    bn(w.peraw, 64, 16, P[TR+15], P[TR+14], w.bnm, w.bnr, 1, s);   // pos_bn+relu
    conv(w.peraw, P[TR+13], P[TR+12], w.pe, 64, 64, NK, 0, s);     // pos2
    ew_add4<<<cdiv(E64/4, 256), 256, 0, s>>>(w.qk, w.qk, w.pe, E64/4);           // qk+pe
    badd_group4<<<cdiv(E64/4, 256), 256, 0, s>>>(w.peraw, w.vf, w.pe, 64, E64/4); // val
    conv(w.qk, P[TR+1], P[TR+0], w.a1, 64, 256, NK, 0, s);         // attn1
    bn(w.a1, 256, 16, P[TR+5], P[TR+4], w.bnm, w.bnr, 1, s);       // attn_bn+relu
    conv(w.a1, P[TR+3], P[TR+2], w.pe, 256, 64, NK, 0, s);         // attn2 (reuse pe)
    softmax16<<<cdiv(64L * NB_, 256), 256, 0, s>>>(w.pe, 64L * NB_);
    attn_agg<<<cdiv(64L * NB_, 256), 256, 0, s>>>(w.agg, w.pe, w.peraw, 64L * NB_);
    conv(w.agg, P[TR+7], P[TR+6], feat, 64, 128, N_, 0, s);        // end
    ew_add4<<<cdiv(32L * NB_, 256), 256, 0, s>>>(feat, feat, w.f3, 32L * NB_);   // + identity

    // ---- RFA (GRU gate), stage 2 only ----
    if (with_rfa && pre_h) {
        const long E128_4 = 32L * NB_;      // (B,128,N)/4
        concat256_4<<<cdiv(2*E128_4, 256), 256, 0, s>>>(w.f2, feat, pre_h, 0, 2*E128_4);
        conv(w.f2, P[29], P[28], w.zb, 256, 128, N_, 0, s);        // z conv
        bn(w.zb, 128, 12, P[27], P[26], w.bnm, w.bnr, 2, s);       // sigmoid
        conv(w.f2, P[25], P[24], w.rb, 256, 128, N_, 0, s);        // r conv
        bn(w.rb, 128, 12, P[23], P[22], w.bnm, w.bnr, 2, s);       // sigmoid
        ew_mul4<<<cdiv(E128_4, 256), 256, 0, s>>>(w.f1, w.rb, feat, E128_4);     // r*cur
        concat256_4<<<cdiv(2*E128_4, 256), 256, 0, s>>>(w.t256, w.f1, pre_h, 0, 2*E128_4);
        conv(w.t256, P[21], P[20], w.hhb, 256, 128, N_, 0, s);     // h conv
        bn(w.hhb, 128, 12, P[19], P[18], w.bnm, w.bnr, 1, s);      // relu
        gru_mix4<<<cdiv(E128_4, 256), 256, 0, s>>>(feat, w.zb, w.hhb, E128_4);
    }

    // ---- GCR offset head ----
    conv(feat, P[1], P[0], w.g64, 128, 64, N_, 1, s);
    conv(w.g64, P[3], P[2], w.g16, 64, 16, N_, 1, s);
    conv(w.g16, P[5], P[4], w.g1,  16, 1,  N_, 0, s);
    conv(feat, P[7], P[6], w.g64, 128, 64, N_, 1, s);              // reuse g64
    conv(w.g64, P[9], P[8], w.r3,  64, 3,  N_, 0, s);
    gcr_apply<<<cdiv((long)NB_, 256), 256, 0, s>>>(pts_out, pts, w.g1, w.r3, 1.0f);
}

extern "C" void kernel_launch(void* const* d_in, const int* in_sizes, int n_in,
                              void* d_out, int out_size, void* d_ws, size_t ws_size,
                              hipStream_t stream)
{
    (void)in_sizes; (void)n_in; (void)out_size; (void)ws_size;
    const float* points = (const float*)d_in[0];
    const float* const* P1 = (const float* const*)(d_in + 1);   // 40 leaves
    const float* const* P2 = (const float* const*)(d_in + 41);  // 52 leaves

    char* base = (char*)d_ws;
    size_t off = 0;
    auto alloc = [&](size_t nfl) -> float* {
        float* p = (float*)(base + off);
        off += ((nfl * sizeof(float) + 255) / 256) * 256;
        return p;
    };

    WS w;
    const size_t NK = (size_t)N_ * 16;
    w.t64    = alloc(64UL  * NB_);
    w.f1     = alloc(128UL * NB_);
    w.gmax   = alloc(128UL * B_);
    w.f2     = alloc(256UL * NB_);
    w.t256   = alloc(256UL * NB_);
    w.f3     = alloc(128UL * NB_);
    w.x64    = alloc(64UL  * NB_);
    w.keyf   = alloc(64UL  * NB_);
    w.qf     = alloc(64UL  * NB_);
    w.vf     = alloc(64UL  * NB_);
    w.posrel = alloc(3UL   * B_ * NK);
    w.peraw  = alloc(64UL  * B_ * NK);   // pos1 out, later "val"
    w.pe     = alloc(64UL  * B_ * NK);   // pos_emb, later attn
    w.qk     = alloc(64UL  * B_ * NK);
    w.a1     = alloc(256UL * B_ * NK);
    w.agg    = alloc(64UL  * NB_);
    w.g64    = alloc(64UL  * NB_);
    w.g16    = alloc(16UL  * NB_);
    w.g1     = alloc(1UL   * NB_);
    w.r3     = alloc(3UL   * NB_);
    w.zb     = alloc(128UL * NB_);
    w.rb     = alloc(128UL * NB_);
    w.hhb    = alloc(128UL * NB_);
    w.bnm    = alloc(256);
    w.bnr    = alloc(256);
    float* feat1 = alloc(128UL * NB_);
    float* feat2 = alloc(128UL * NB_);
    float* p1    = alloc(3UL * NB_);
    w.idx = (int*)alloc((size_t)NB_ * 16);

    // stage 1: points -> p1 (+ hidden feat1)
    run_stage(points, p1, feat1, nullptr, P1, 0, w, stream);
    // stage 2: p1 -> d_out, GRU-fused with feat1
    run_stage(p1, (float*)d_out, feat2, feat1, P2, 1, w, stream);
}
